// MulitpleHeads_72052371358028
// MI455X (gfx1250) — matrix operand order repaired and checked by reading, not compiled, not run on hardware
//
#include <hip/hip_runtime.h>

// MI455X / gfx1250 fused multi-head attention.
// B=4096, T=32, H=16, HS=32, E=512.
// 2 batch elements per workgroup (M=64 GEMMs) to halve L2 weight traffic.
// All matmuls via v_wmma_f32_16x16x32_f16; x staged via global_load_async_to_lds.

typedef __attribute__((ext_vector_type(16))) _Float16 v16h;
typedef __attribute__((ext_vector_type(8)))  _Float16 v8h;
typedef __attribute__((ext_vector_type(8)))  float    v8f;

#define WMMA_F16(A, Bm, C) \
  __builtin_amdgcn_wmma_f32_16x16x32_f16(false, (A), false, (Bm), (short)0, (C), false, false)

// ---------------------------------------------------------------------------
// Prologue: convert weights to f16 in [N][K] layout (row = N, contiguous K).
//   ws[0     ..) : WqT [16][32][512] f16   (WT[h][d][e] = Wq[h][d][e])
//   ws[256K  ..) : WkT
//   ws[512K  ..) : WvT
//   ws[768K  ..) : WpT [512][512]    f16   (WpT[n][k]   = Wp[n][k])
// ---------------------------------------------------------------------------
__global__ void convert_weights(const float* __restrict__ Wq,
                                const float* __restrict__ Wk,
                                const float* __restrict__ Wv,
                                const float* __restrict__ Wp,
                                _Float16* __restrict__ ws) {
  const int idx = blockIdx.x * 256 + threadIdx.x;   // 0 .. 4*262144-1
  const int region = idx >> 18;
  const int r = idx & 262143;
  float v;
  if (region < 3) {
    const float* W = (region == 0) ? Wq : (region == 1) ? Wk : Wv;
    const int h  = r >> 14;          // 16384 = 512*32
    const int rr = r & 16383;
    const int e  = rr & 511;
    const int d  = rr >> 9;
    v = W[(h * 32 + d) * 512 + e];
  } else {
    const int k = r >> 9;
    const int n = r & 511;
    v = Wp[k * 512 + n];
  }
  ws[idx] = (_Float16)v;
}

// ---------------------------------------------------------------------------
// A-operand loader: 16x32 f16 tile from LDS, CDNA5 layout.
// lane L: row M = r0 + (L%16); e<8 -> K = k0 + 8*(L/16) + e,
//                              e>=8 -> K = k0 + 16 + 8*(L/16) + (e-8).
// `row` already includes (L%16); hl = L/16. Row base must be 16B aligned.
// ---------------------------------------------------------------------------
__device__ __forceinline__ v16h load_a(const _Float16* base, int stride,
                                       int row, int k0, int hl) {
  const _Float16* p = base + row * stride + k0 + 8 * hl;
  v8h lo = *(const v8h*)(p);
  v8h hi = *(const v8h*)(p + 16);
  v16h a;
#pragma unroll
  for (int i = 0; i < 8; ++i) { a[i] = lo[i]; a[i + 8] = hi[i]; }
  return a;
}

// ---------------------------------------------------------------------------
// Fused attention: one workgroup (256 thr = 8 wave32) per 2 batch elements.
// LDS (dynamic, 317440 B):
//   xs  [64][520] f16      x tiles (row = b*32+t)      @      0   (aliased by Z)
//   Z   [64][520] f16      attn out (row = b*32+d)     @      0
//   qs  [32][32][32] f16   q per (b,h) instance        @  66560   (aliased by stage)
//   kT  [32][32][32] f16   k per instance [t][d]       @ 132096
//   vs  [32][32][32] f16   v^T per instance [d][t]     @ 197632
//   wei [8][32][40] f16    softmax weights per wave    @ 263168
//   sc  [8][32][33] f32    raw scores per wave         @ 283648
//   stage [2][32][512] f32 async x staging             @  66560
// ---------------------------------------------------------------------------
__global__ __launch_bounds__(256, 1)
void mha_fused(const float* __restrict__ x,
               const _Float16* __restrict__ WT,    // [3][16][32][512] f16
               const _Float16* __restrict__ WpT,   // [512][512] f16
               const float* __restrict__ bp,       // [512] f32
               float* __restrict__ out) {          // [B][32][512] f32
  extern __shared__ char smem[];
  _Float16* xs    = (_Float16*)(smem);
  _Float16* Z     = (_Float16*)(smem);
  _Float16* qs    = (_Float16*)(smem + 66560);
  _Float16* kT    = (_Float16*)(smem + 132096);
  _Float16* vs    = (_Float16*)(smem + 197632);
  _Float16* wei   = (_Float16*)(smem + 263168);
  float*    sc    = (float*)(smem + 283648);
  float*    stage = (float*)(smem + 66560);

  const int bid  = blockIdx.x;        // handles batches 2*bid, 2*bid+1
  const int tid  = threadIdx.x;
  const int wave = tid >> 5;
  const int lane = tid & 31;
  const int ln   = lane & 15;
  const int hl   = lane >> 4;

  // ---- Phase 0a: async-copy both x tiles (128 KB f32) global -> LDS ----
  {
    const uint64_t gbase = (uint64_t)(uintptr_t)(x + (size_t)bid * (2 * 32 * 512));
    const uint32_t lbase = (uint32_t)(uintptr_t)(void*)stage;  // flat low 32b = LDS offset
#pragma unroll
    for (int i = 0; i < 32; ++i) {
      const uint32_t boff  = (uint32_t)(tid + i * 256) * 16u;  // 16B per lane
      const uint32_t laddr = lbase + boff;
      const uint64_t gaddr = gbase + boff;
      asm volatile("global_load_async_to_lds_b128 %0, %1, off"
                   :: "v"(laddr), "v"(gaddr) : "memory");
    }
    asm volatile("s_wait_asynccnt 0x0" ::: "memory");
  }
  __syncthreads();

  // ---- Phase 0b: convert staged f32 -> f16 xs[64][520] ----
#pragma unroll
  for (int i = 0; i < 32; ++i) {
    const int idx4 = tid + i * 256;
    float4 f = ((const float4*)stage)[idx4];
    const int flat = idx4 * 4;
    const int row = flat >> 9;          // 0..63 (= b*32 + t)
    const int col = flat & 511;
    _Float16* d = xs + row * 520 + col;
    d[0] = (_Float16)f.x; d[1] = (_Float16)f.y;
    d[2] = (_Float16)f.z; d[3] = (_Float16)f.w;
  }
  __syncthreads();

  // ---- Phase 1: q/k/v projections. [64 x 512] @ [512 x 32] per (matrix,head).
  // 48 pairs, 6 per wave; B streamed once, reused by 4 M-tiles. ----
  for (int p = 0; p < 6; ++p) {
    const int pi = wave * 6 + p;        // pi = m*16 + h
    const int m  = pi >> 4;
    const int h  = pi & 15;
    const _Float16* Wh = WT + (size_t)pi * (512 * 32);
    for (int nt = 0; nt < 2; ++nt) {
      v8f acc[4] = {};
      for (int k0 = 0; k0 < 512; k0 += 32) {
        v16h bm = load_a(Wh, 512, nt * 16 + ln, k0, hl);
#pragma unroll
        for (int mt = 0; mt < 4; ++mt) {
          v16h a = load_a(xs, 520, mt * 16 + ln, k0, hl);
          acc[mt] = WMMA_F16(a, bm, acc[mt]);
        }
      }
      const int d = nt * 16 + ln;
#pragma unroll
      for (int mt = 0; mt < 4; ++mt) {
        const int inst = (mt >> 1) * 16 + h;   // batch-in-pair * 16 + head
#pragma unroll
        for (int r = 0; r < 8; ++r) {
          const int t = ((mt & 1) * 16) + r + 8 * hl;   // 0..31 within batch
          const _Float16 val = (_Float16)acc[mt][r];
          if (m == 0)      qs[inst * 1024 + t * 32 + d] = val;
          else if (m == 1) kT[inst * 1024 + t * 32 + d] = val;
          else             vs[inst * 1024 + d * 32 + t] = val;  // transposed
        }
      }
    }
  }
  __syncthreads();

  // ---- Phase 2: attention, 4 (batch,head) instances per wave ----
  const float scale = 0.17677669529663687f;  // 1/sqrt(32)
  for (int ii = 0; ii < 4; ++ii) {
    const int inst = wave * 4 + ii;          // = b*16 + h
    const int bb = inst >> 4;
    const int h  = inst & 15;
    const _Float16* qh = qs + inst * 1024;
    const _Float16* kh = kT + inst * 1024;
    const _Float16* vh = vs + inst * 1024;
    // scores = q @ k^T, scale + causal mask -> sc
#pragma unroll
    for (int mt = 0; mt < 2; ++mt)
#pragma unroll
      for (int nt = 0; nt < 2; ++nt) {
        v16h a  = load_a(qh, 32, mt * 16 + ln, 0, hl);
        v16h bm = load_a(kh, 32, nt * 16 + ln, 0, hl);
        v8f acc = {};
        acc = WMMA_F16(a, bm, acc);
#pragma unroll
        for (int r = 0; r < 8; ++r) {
          const int t = mt * 16 + r + 8 * hl;
          const int s = nt * 16 + ln;
          sc[wave * 1056 + t * 33 + s] = (s <= t) ? acc[r] * scale : -3.0e38f;
        }
      }
    // softmax: lane L owns row t = L
    {
      const float* rowp = sc + wave * 1056 + lane * 33;
      float mx = rowp[0];
#pragma unroll
      for (int s = 1; s < 32; ++s) mx = fmaxf(mx, rowp[s]);
      float ev[32];
      float sum = 0.f;
#pragma unroll
      for (int s = 0; s < 32; ++s) { ev[s] = __expf(rowp[s] - mx); sum += ev[s]; }
      const float inv = 1.0f / sum;
      _Float16* wrow = wei + wave * 1280 + lane * 40;
#pragma unroll
      for (int s = 0; s < 32; ++s) wrow[s] = (_Float16)(ev[s] * inv);
    }
    // out_h = wei @ v ; store transposed: Z[b*32 + d][h*32 + t]
#pragma unroll
    for (int mt = 0; mt < 2; ++mt)
#pragma unroll
      for (int nt = 0; nt < 2; ++nt) {
        v16h a  = load_a(wei + wave * 1280, 40, mt * 16 + ln, 0, hl);
        v16h bm = load_a(vh, 32, nt * 16 + ln, 0, hl);
        v8f acc = {};
        acc = WMMA_F16(a, bm, acc);
#pragma unroll
        for (int r = 0; r < 8; ++r) {
          const int t = mt * 16 + r + 8 * hl;
          const int d = nt * 16 + ln;
          Z[(bb * 32 + d) * 520 + h * 32 + t] = (_Float16)acc[r];
        }
      }
  }
  __syncthreads();

  // ---- Phase 3: Y = Z @ WpT + bp ; [64 x 512] @ [512 x 512], 4 N-tiles/wave ----
  float* outb = out + (size_t)bid * (64 * 512);
  for (int q = 0; q < 4; ++q) {
    const int nt = wave * 4 + q;
    v8f acc[4] = {};
    for (int k0 = 0; k0 < 512; k0 += 32) {
      v16h bm = load_a(WpT, 512, nt * 16 + ln, k0, hl);
#pragma unroll
      for (int mt = 0; mt < 4; ++mt) {
        v16h a = load_a(Z, 520, mt * 16 + ln, k0, hl);
        acc[mt] = WMMA_F16(a, bm, acc[mt]);
      }
    }
    const int col = nt * 16 + ln;
    const float bias = bp[col];
#pragma unroll
    for (int mt = 0; mt < 4; ++mt)
#pragma unroll
      for (int r = 0; r < 8; ++r) {
        const int row = mt * 16 + r + 8 * hl;   // = b*32 + d
        outb[row * 512 + col] = acc[mt][r] + bias;
      }
  }
}

// ---------------------------------------------------------------------------
extern "C" void kernel_launch(void* const* d_in, const int* in_sizes, int n_in,
                              void* d_out, int out_size, void* d_ws, size_t ws_size,
                              hipStream_t stream) {
  const float* x  = (const float*)d_in[0];
  const float* Wq = (const float*)d_in[1];
  const float* Wk = (const float*)d_in[2];
  const float* Wv = (const float*)d_in[3];
  const float* Wp = (const float*)d_in[4];
  const float* bp = (const float*)d_in[5];

  _Float16* ws = (_Float16*)d_ws;          // 1,048,576 halfs = 2 MB

  convert_weights<<<4096, 256, 0, stream>>>(Wq, Wk, Wv, Wp, ws);

  const _Float16* WT  = ws;                // [3][16][32][512]
  const _Float16* WpT = ws + 786432;       // [512][512]

  constexpr int SMEM_BYTES = 317440;       // see layout in kernel
  mha_fused<<<2048, 256, SMEM_BYTES, stream>>>(x, WT, WpT, bp, (float*)d_out);
}
